// LC_CRF_44255343018665
// MI455X (gfx1250) — matrix operand order repaired and checked by reading, not compiled, not run on hardware
//
#include <hip/hip_runtime.h>
#include <hip/hip_bf16.h>

// Problem constants (from reference): B=512, S=1024, L=64
#define kB 512
#define kS 1024
#define kL 64
#define kPstr 72                      // padded LDS row stride (bf16 elems), 144B rows
#define kLog2e 1.4426950408889634f
#define kLn2   0.6931471805599453f

typedef __attribute__((ext_vector_type(16))) __bf16 v16bf;
typedef __attribute__((ext_vector_type(8)))  __bf16 v8bf;
typedef __attribute__((ext_vector_type(8)))  float  v8f;

// XOR-butterfly add across the 16 lanes of a half-wave, pure VALU
// (v_permlane16_b32 with full permutation tables; no LDS traffic).
__device__ __forceinline__ float permlane16_add(float v, unsigned s0, unsigned s1) {
  const int i = __builtin_bit_cast(int, v);
  const int p = __builtin_amdgcn_permlane16(i, i, s0, s1, false, false);
  return v + __builtin_bit_cast(float, p);
}
__device__ __forceinline__ float halfwave_rowsum(float v) {
  v = permlane16_add(v, 0x67452301u, 0xEFCDAB89u); // xor 1
  v = permlane16_add(v, 0x54761032u, 0xDCFE98BAu); // xor 2
  v = permlane16_add(v, 0x32107654u, 0xBA98FEDCu); // xor 4
  v = permlane16_add(v, 0xFEDCBA98u, 0x76543210u); // xor 8
  return v;
}

// ---------------------------------------------------------------------------
// Forward scan: log-partition in scaled-probability domain via bf16 WMMA.
// Block = 128 threads (4 waves); block owns 16 batch rows, wave w owns label
// columns 16w..16w+15. E = exp(T) stays in registers as WMMA B-fragments.
// P tile double-buffers through LDS (one barrier/step). Row renormalization
// every 4 steps, applied lazily one step later (scaling commutes with the
// linear recursion), so renorm adds no extra barrier.
// ---------------------------------------------------------------------------
__global__ __launch_bounds__(128, 1) void crf_partition_wmma(
    const float* __restrict__ em, const float* __restrict__ mask,
    const float* __restrict__ startT, const float* __restrict__ endT,
    const float* __restrict__ trans, float* __restrict__ partition)
{
  __shared__ __attribute__((aligned(16))) __bf16 Pt[2][16 * kPstr];
  __shared__ __attribute__((aligned(16))) float  part[16 * 4];

  const int tid  = threadIdx.x;
  const int w    = tid >> 5;          // wave id 0..3 -> N tile
  const int lane = tid & 31;
  const int li   = lane & 15;
  const int half = lane >> 4;
  const int j    = w * 16 + li;       // label column this lane covers in D
  const int rb   = half * 8;          // first batch row of this lane's D elems
  const int b0   = blockIdx.x * 16;

  // Static B fragments: E[i][j] = exp(T[i][j]) (bf16 B layout: lane li holds
  // column j; lanes 0-15 K=0..7,16..23, lanes 16-31 K=8..15,24..31 per 32-row fragment).
  v16bf B0, B1;
#pragma unroll
  for (int e = 0; e < 16; ++e) {
    const int i0 = (e < 8) ? (8 * half + e) : (16 + 8 * half + (e - 8));
    B0[e] = (__bf16)__builtin_amdgcn_exp2f(kLog2e * trans[i0 * kL + j]);
    B1[e] = (__bf16)__builtin_amdgcn_exp2f(kLog2e * trans[(32 + i0) * kL + j]);
  }

  float s2[8];                        // per-row log2 scale accumulator
#pragma unroll
  for (int e = 0; e < 8; ++e) s2[e] = 0.f;

  const size_t rowStride = (size_t)kS * kL;
  const float* emLane = em   + (size_t)(b0 + rb) * rowStride + j;
  const float* mkLane = mask + (size_t)(b0 + rb) * kS;

  // --- t = 0: alpha0 = start + em[:,0] (probability domain, unnormalized)
  float val[8], prev[8];
  const float st = startT[j];
#pragma unroll
  for (int e = 0; e < 8; ++e) {
    val[e]  = __builtin_amdgcn_exp2f(kLog2e * (st + emLane[e * rowStride]));
    prev[e] = val[e];
    Pt[0][(rb + e) * kPstr + j] = (__bf16)val[e];
  }

  float emn[8], mkn[8];               // register prefetch of t = 1
#pragma unroll
  for (int e = 0; e < 8; ++e) {
    emn[e] = emLane[e * rowStride + kL];
    mkn[e] = mkLane[e * kS + 1];
  }
  __syncthreads();

  for (int t = 1; t < kS; ++t) {
    // Lazy renormalization: partials for step t-1 were stored before the
    // barrier iff (t-1)%4==3; fold 1/rowsum into this step's inputs.
    float pinv[8];
    if (((t - 1) & 3) == 3) {
#pragma unroll
      for (int e = 0; e < 8; ++e) {
        const float4 p  = *(const float4*)&part[(rb + e) * 4];
        const float tot = (p.x + p.y) + (p.z + p.w);
        s2[e]  += __builtin_amdgcn_logf(tot);   // v_log_f32 = log2
        pinv[e] = __builtin_amdgcn_rcpf(tot);
      }
    } else {
#pragma unroll
      for (int e = 0; e < 8; ++e) pinv[e] = 1.f;
    }

    // A fragments: row li of P(t-1) from the read buffer.
    const __bf16* rowp = &Pt[(t - 1) & 1][li * kPstr];
    const v8bf a0 = *(const v8bf*)(rowp +       8 * half);
    const v8bf a1 = *(const v8bf*)(rowp + 16 +  8 * half);
    const v8bf a2 = *(const v8bf*)(rowp + 32 +  8 * half);
    const v8bf a3 = *(const v8bf*)(rowp + 48 +  8 * half);
    const v16bf A0 = __builtin_shufflevector(a0, a1, 0,1,2,3,4,5,6,7,8,9,10,11,12,13,14,15);
    const v16bf A1 = __builtin_shufflevector(a2, a3, 0,1,2,3,4,5,6,7,8,9,10,11,12,13,14,15);

    float emc[8], mkc[8];
#pragma unroll
    for (int e = 0; e < 8; ++e) { emc[e] = emn[e]; mkc[e] = mkn[e]; }
    if (t + 1 < kS) {                 // register prefetch of t+1
#pragma unroll
      for (int e = 0; e < 8; ++e) {
        emn[e] = emLane[e * rowStride + (size_t)(t + 1) * kL];
        mkn[e] = mkLane[e * kS + (t + 1)];
      }
    }
    if (t + 8 < kS) __builtin_prefetch(emLane + (size_t)(t + 8) * kL, 0, 1);

    // new = (P x E) * exp(em): f32 accumulate, K=64 via two k32 bf16 WMMAs.
    v8f c = {};
    c = __builtin_amdgcn_wmma_f32_16x16x32_bf16(false, A0, false, B0, (short)0, c, false, false);
    c = __builtin_amdgcn_wmma_f32_16x16x32_bf16(false, A1, false, B1, (short)0, c, false, false);

#pragma unroll
    for (int e = 0; e < 8; ++e) {
      const float nv = c[e] * __builtin_amdgcn_exp2f(kLog2e * emc[e]);
      val[e]  = pinv[e] * (prev[e] + mkc[e] * (nv - prev[e]));  // mask in {0,1}
      prev[e] = val[e];
    }

    // Renorm bookkeeping every 4th step (incl. t = S-1 = 1023): VALU-only
    // butterfly, partials into LDS before the regular barrier.
    if ((t & 3) == 3) {
      float r[8];
#pragma unroll
      for (int e = 0; e < 8; ++e) r[e] = halfwave_rowsum(val[e]);
      if (li == 0) {
#pragma unroll
        for (int e = 0; e < 8; ++e) part[(rb + e) * 4 + w] = r[e];
      }
    }

#pragma unroll
    for (int e = 0; e < 8; ++e)
      Pt[t & 1][(rb + e) * kPstr + j] = (__bf16)val[e];
    __syncthreads();
  }

  // --- Finalize: t=1023 partials are in `part` (post-barrier). Fold the last
  // normalizer, weight by exp(end), reduce, emit per-batch partition.
  const float endw = __builtin_amdgcn_exp2f(kLog2e * endT[j]);
  float fv[8];
#pragma unroll
  for (int e = 0; e < 8; ++e) {
    const float4 p  = *(const float4*)&part[(rb + e) * 4];
    const float tot = (p.x + p.y) + (p.z + p.w);
    s2[e] += __builtin_amdgcn_logf(tot);
    fv[e]  = val[e] * __builtin_amdgcn_rcpf(tot) * endw;
  }
  __syncthreads();                    // everyone done reading `part`
  float r[8];
#pragma unroll
  for (int e = 0; e < 8; ++e) r[e] = halfwave_rowsum(fv[e]);
  if (li == 0) {
#pragma unroll
    for (int e = 0; e < 8; ++e) part[(rb + e) * 4 + w] = r[e];
  }
  __syncthreads();
  if (w == 0 && li == 0) {
#pragma unroll
    for (int e = 0; e < 8; ++e) {
      const float4 p  = *(const float4*)&part[(rb + e) * 4];
      const float tot = (p.x + p.y) + (p.z + p.w);
      partition[b0 + rb + e] = kLn2 * (s2[e] + __builtin_amdgcn_logf(tot));
    }
  }
}

// ---------------------------------------------------------------------------
// Gold path scores + final scalar reduction: out = sum_b(partition_b - score_b)
// ---------------------------------------------------------------------------
__global__ __launch_bounds__(512) void crf_gold_reduce(
    const float* __restrict__ em, const int* __restrict__ tags,
    const float* __restrict__ mask, const float* __restrict__ startT,
    const float* __restrict__ endT, const float* __restrict__ trans,
    const float* __restrict__ partition, float* __restrict__ out)
{
  const int b = threadIdx.x;
  const size_t base = (size_t)b * kS;
  int prev = tags[base];
  float score = startT[prev] + em[base * kL + prev];
  float msum  = mask[base];
  for (int s = 1; s < kS; ++s) {
    const int tg  = tags[base + s];
    const float m = mask[base + s];
    score += m * (em[(base + s) * kL + tg] + trans[prev * kL + tg]);
    msum  += m;
    prev = tg;
  }
  const int last = (int)msum - 1;
  score += endT[tags[base + last]];

  __shared__ float red[512];
  red[b] = partition[b] - score;
  for (int off = 256; off > 0; off >>= 1) {
    __syncthreads();
    if (b < off) red[b] += red[b + off];
  }
  __syncthreads();
  if (b == 0) out[0] = red[0];
}

extern "C" void kernel_launch(void* const* d_in, const int* in_sizes, int n_in,
                              void* d_out, int out_size, void* d_ws, size_t ws_size,
                              hipStream_t stream) {
  (void)in_sizes; (void)n_in; (void)out_size; (void)ws_size;
  const float* em    = (const float*)d_in[0];
  const int*   tags  = (const int*)d_in[1];
  const float* mask  = (const float*)d_in[2];
  const float* stT   = (const float*)d_in[3];
  const float* enT   = (const float*)d_in[4];
  const float* trT   = (const float*)d_in[5];
  float* partition   = (float*)d_ws;        // 512 floats of scratch

  crf_partition_wmma<<<kB / 16, 128, 0, stream>>>(em, mask, stT, enT, trT, partition);
  crf_gold_reduce<<<1, 512, 0, stream>>>(em, tags, mask, stT, enT, trT, partition,
                                         (float*)d_out);
}
